// SpatialGatedBlock_21397527068865
// MI455X (gfx1250) — matrix operand order repaired and checked by reading, not compiled, not run on hardware
//
#include <hip/hip_runtime.h>
#include <math.h>

#define DIM 128
#define ZSLICES 32

typedef __attribute__((ext_vector_type(16))) _Float16 v16h;
typedef __attribute__((ext_vector_type(8)))  _Float16 v8h;
typedef __attribute__((ext_vector_type(8)))  float    v8f;

// ---------------------------------------------------------------------------
// K0: per-row L2 norm; emit f16 copies of x (Xh) and normalized rows (Yh),
// both row-major [N][DIM]. One wave32 per row, 4 channels per lane.
// ---------------------------------------------------------------------------
__global__ void prep_kernel(const float* __restrict__ x,
                            _Float16* __restrict__ Xh,
                            _Float16* __restrict__ Yh, int n) {
  int wave = (blockIdx.x * blockDim.x + threadIdx.x) >> 5;
  int lane = threadIdx.x & 31;
  if (wave >= n) return;
  const float* xr = x + (size_t)wave * DIM;
  float v0 = xr[lane], v1 = xr[lane + 32], v2 = xr[lane + 64], v3 = xr[lane + 96];
  float s = v0 * v0 + v1 * v1 + v2 * v2 + v3 * v3;
  #pragma unroll
  for (int off = 16; off > 0; off >>= 1) s += __shfl_xor(s, off, 32);
  float nrm  = sqrtf(s);
  float rinv = nrm > 0.f ? 1.f / nrm : 0.f;
  _Float16* xo = Xh + (size_t)wave * DIM;
  _Float16* yo = Yh + (size_t)wave * DIM;
  xo[lane]      = (_Float16)v0;  yo[lane]      = (_Float16)(v0 * rinv);
  xo[lane + 32] = (_Float16)v1;  yo[lane + 32] = (_Float16)(v1 * rinv);
  xo[lane + 64] = (_Float16)v2;  yo[lane + 64] = (_Float16)(v2 * rinv);
  xo[lane + 96] = (_Float16)v3;  yo[lane + 96] = (_Float16)(v3 * rinv);
}

// ---------------------------------------------------------------------------
// LDS-tiled f16 transpose: in [N][DIM] -> out [DIM][N]. 64 rows per block.
// Row pitch 130 halfs -> 260B stride -> conflict-free column reads.
// ---------------------------------------------------------------------------
#define TPAD 130
__global__ void tr_kernel(const _Float16* __restrict__ in,
                          _Float16* __restrict__ out, int n) {
  __shared__ _Float16 lds[64 * TPAD];
  int t = threadIdx.x;
  int i0 = blockIdx.x * 64;
  int r = t >> 2, q = t & 3;                       // 4 threads per row, 32 ch each
  const unsigned* src32 = (const unsigned*)(in + (size_t)(i0 + r) * DIM + q * 32);
  unsigned* dst32 = (unsigned*)&lds[r * TPAD + q * 32];
  #pragma unroll
  for (int j = 0; j < 16; ++j) dst32[j] = src32[j];
  __syncthreads();
  int il = t & 63, cg = t >> 6;                    // 64 i's coalesced per store
  #pragma unroll
  for (int j = 0; j < 32; ++j) {
    int c = cg * 32 + j;
    out[(size_t)c * n + i0 + il] = lds[il * TPAD + c];
  }
}

// W [DIM][DIM] f32 -> WhT [DIM][DIM] f16 transposed (K-contiguous B operand).
__global__ void convwt_kernel(const float* __restrict__ W, _Float16* __restrict__ WhT) {
  int t = blockIdx.x * blockDim.x + threadIdx.x;   // DIM*DIM
  int k = t >> 7, c = t & (DIM - 1);
  WhT[c * DIM + k] = (_Float16)W[t];
}

// ---------------------------------------------------------------------------
// Degree over target nodes (L2 atomics), then dis = rsqrt(deg + 1) (self loop).
// ---------------------------------------------------------------------------
__global__ void deg_kernel(const int* __restrict__ col, float* __restrict__ deg, int E) {
  int e = blockIdx.x * blockDim.x + threadIdx.x;
  if (e < E) atomicAdd(&deg[col[e]], 1.0f);
}

__global__ void dis_kernel(float* __restrict__ deg, int n) {
  int i = blockIdx.x * blockDim.x + threadIdx.x;
  if (i < n) deg[i] = rsqrtf(deg[i] + 1.0f);
}

// ---------------------------------------------------------------------------
// WMMA K-loop. Wave32 layouts (C/D per CDNA5 ISA §7.12.2; A/B use contiguous k, not the ISA table order):
//  A (16x32 f16): M=lane&15; lane-half hi: halfs [k0+hi*16 .. +15]
//                 -> one contiguous v16h load (same k order as B).
//  B (32x16 f16): N=lane&15; halfs [k0+hi*16 .. +15] of the K-contiguous
//                 (transposed) B row -> one contiguous v16h load.
//  C/D (16x16 f32): N=lane&15; VGPR v -> M = v + 8*hi.
// arow/brow are the per-lane rows (already offset by M / N).
// ---------------------------------------------------------------------------
template <bool PF>
__device__ __forceinline__ v8f wmma_kloop(const _Float16* __restrict__ arow,
                                          const _Float16* __restrict__ brow,
                                          int kbeg, int kend, v8f acc) {
  int hi = (threadIdx.x >> 4) & 1;
  for (int k0 = kbeg; k0 < kend; k0 += 32) {
    if (PF) {
      __builtin_prefetch(arow + k0 + 128, 0, 1);
      __builtin_prefetch(brow + k0 + 128, 0, 1);
    }
    v16h a   = *(const v16h*)(arow + k0 + hi * 16);
    v16h b   = *(const v16h*)(brow + k0 + hi * 16);
    acc = __builtin_amdgcn_wmma_f32_16x16x32_f16(false, a, false, b,
                                                 (short)0, acc, false, false);
  }
  return acc;
}

// ---------------------------------------------------------------------------
// K1: Z = Y^T X (128x128, K = N). K split into ZSLICES; f32 partials to ws.
// A row m of Y^T == YhT[m][*]; B col n of X == XhT[n][*] (both K-contiguous).
// ---------------------------------------------------------------------------
__global__ void ztx_gemm_kernel(const _Float16* __restrict__ YhT,
                                const _Float16* __restrict__ XhT,
                                float* __restrict__ Zpart, int n) {
  int wave = (blockIdx.x * blockDim.x + threadIdx.x) >> 5;
  int lane = threadIdx.x & 31, Ml = lane & 15, hi = lane >> 4;
  int slice = wave & (ZSLICES - 1);
  int tile  = wave >> 5;                           // 0..63
  if (tile >= (DIM / 16) * (DIM / 16)) return;
  int m0 = (tile >> 3) * 16, n0 = (tile & 7) * 16;
  int klen = n / ZSLICES;
  int kbeg = slice * klen;
  const _Float16* arow = YhT + (size_t)(m0 + Ml) * n;
  const _Float16* brow = XhT + (size_t)(n0 + Ml) * n;
  v8f acc = {};
  acc = wmma_kloop<true>(arow, brow, kbeg, kbeg + klen, acc);
  float* zp = Zpart + (size_t)slice * DIM * DIM;
  #pragma unroll
  for (int v = 0; v < 8; ++v)
    zp[(m0 + v + 8 * hi) * DIM + n0 + Ml] = acc[v];
}

// Deterministic reduction of the K-slices; emit Z transposed (ZhT[n][k]).
__global__ void zreduce_kernel(const float* __restrict__ Zpart,
                               _Float16* __restrict__ ZhT) {
  int i = blockIdx.x * blockDim.x + threadIdx.x;   // DIM*DIM
  float s = 0.f;
  #pragma unroll
  for (int sl = 0; sl < ZSLICES; ++sl) s += Zpart[sl * DIM * DIM + i];
  int m = i >> 7, c = i & (DIM - 1);
  ZhT[c * DIM + m] = (_Float16)s;
}

// ---------------------------------------------------------------------------
// K2: xw = X W via WMMA ([N,128] x [128,128]); f32 result for the scatter.
// ---------------------------------------------------------------------------
__global__ void xw_gemm_kernel(const _Float16* __restrict__ Xh,
                               const _Float16* __restrict__ WhT,
                               float* __restrict__ xw, int tiles) {
  int wave = (blockIdx.x * blockDim.x + threadIdx.x) >> 5;
  if (wave >= tiles) return;
  int lane = threadIdx.x & 31, Ml = lane & 15, hi = lane >> 4;
  int tm = wave >> 3, tn = wave & 7;
  const _Float16* arow = Xh + (size_t)(tm * 16 + Ml) * DIM;
  const _Float16* brow = WhT + (size_t)(tn * 16 + Ml) * DIM;
  v8f acc = {};
  acc = wmma_kloop<false>(arow, brow, 0, DIM, acc);
  #pragma unroll
  for (int v = 0; v < 8; ++v)
    xw[(size_t)(tm * 16 + v + 8 * hi) * DIM + tn * 16 + Ml] = acc[v];
}

// ---------------------------------------------------------------------------
// K3: edge scatter. One wave per edge; 4 channels per lane; f32 L2 atomics.
// ---------------------------------------------------------------------------
__global__ void scatter_kernel(const int* __restrict__ row, const int* __restrict__ col,
                               const float* __restrict__ dis,
                               const float* __restrict__ xw,
                               float* __restrict__ g, int E) {
  int wave = (blockIdx.x * blockDim.x + threadIdx.x) >> 5;
  int lane = threadIdx.x & 31;
  if (wave >= E) return;
  int r = row[wave], c = col[wave];
  float nrm = dis[r] * dis[c];
  const float* src = xw + (size_t)r * DIM;
  float* dst = g + (size_t)c * DIM;
  #pragma unroll
  for (int j = 0; j < 4; ++j) {
    int ch = lane + 32 * j;
    atomicAdd(&dst[ch], src[ch] * nrm);
  }
}

// K4: self-loop message (norm = dis[i]^2) + bias.
__global__ void self_bias_kernel(const float* __restrict__ xw,
                                 const float* __restrict__ dis,
                                 const float* __restrict__ b,
                                 float* __restrict__ g, int total) {
  int i = blockIdx.x * blockDim.x + threadIdx.x;
  if (i >= total) return;
  int node = i >> 7, ch = i & (DIM - 1);
  float d = dis[node];
  g[i] += xw[i] * d * d + b[ch];
}

// ---------------------------------------------------------------------------
// K5: att_lin = Y Z (WMMA), fused with out = g * sigmoid(att_lin).
// ---------------------------------------------------------------------------
__global__ void att_final_kernel(const _Float16* __restrict__ Yh,
                                 const _Float16* __restrict__ ZhT,
                                 const float* __restrict__ g,
                                 float* __restrict__ out, int tiles) {
  int wave = (blockIdx.x * blockDim.x + threadIdx.x) >> 5;
  if (wave >= tiles) return;
  int lane = threadIdx.x & 31, Ml = lane & 15, hi = lane >> 4;
  int tm = wave >> 3, tn = wave & 7;
  const _Float16* arow = Yh + (size_t)(tm * 16 + Ml) * DIM;
  const _Float16* brow = ZhT + (size_t)(tn * 16 + Ml) * DIM;
  v8f acc = {};
  acc = wmma_kloop<false>(arow, brow, 0, DIM, acc);
  #pragma unroll
  for (int v = 0; v < 8; ++v) {
    size_t idx = (size_t)(tm * 16 + v + 8 * hi) * DIM + tn * 16 + Ml;
    float sgm = 1.0f / (1.0f + expf(-acc[v]));
    out[idx] = g[idx] * sgm;
  }
}

// ---------------------------------------------------------------------------
extern "C" void kernel_launch(void* const* d_in, const int* in_sizes, int n_in,
                              void* d_out, int out_size, void* d_ws, size_t ws_size,
                              hipStream_t stream) {
  const float* x  = (const float*)d_in[0];
  const int*   ei = (const int*)d_in[1];
  const float* W  = (const float*)d_in[2];
  const float* b  = (const float*)d_in[3];
  const int n = in_sizes[0] / DIM;   // 8192
  const int E = in_sizes[1] / 2;     // 262144
  const int* rowp = ei;
  const int* colp = ei + E;

  // workspace layout (~19.0 MB)
  char* ws = (char*)d_ws;
  size_t off = 0;
  _Float16* Xh  = (_Float16*)(ws + off); off += (size_t)n * DIM * 2;
  _Float16* Yh  = (_Float16*)(ws + off); off += (size_t)n * DIM * 2;
  _Float16* XhT = (_Float16*)(ws + off); off += (size_t)n * DIM * 2;
  _Float16* YhT = (_Float16*)(ws + off); off += (size_t)n * DIM * 2;
  _Float16* WhT = (_Float16*)(ws + off); off += (size_t)DIM * DIM * 2;
  _Float16* ZhT = (_Float16*)(ws + off); off += (size_t)DIM * DIM * 2;
  float* Zpart  = (float*)(ws + off);    off += (size_t)ZSLICES * DIM * DIM * 4;
  float* xw     = (float*)(ws + off);    off += (size_t)n * DIM * 4;
  float* g      = (float*)(ws + off);    off += (size_t)n * DIM * 4;
  float* dis    = (float*)(ws + off);    off += (size_t)n * 4;
  float* out    = (float*)d_out;

  hipMemsetAsync(dis, 0, (size_t)n * sizeof(float), stream);
  hipMemsetAsync(g, 0, (size_t)n * DIM * sizeof(float), stream);

  const int tiles = (n / 16) * (DIM / 16);  // 4096 16x16 output tiles

  prep_kernel<<<n / 8, 256, 0, stream>>>(x, Xh, Yh, n);
  tr_kernel<<<n / 64, 256, 0, stream>>>(Xh, XhT, n);
  tr_kernel<<<n / 64, 256, 0, stream>>>(Yh, YhT, n);
  convwt_kernel<<<(DIM * DIM) / 256, 256, 0, stream>>>(W, WhT);
  deg_kernel<<<(E + 255) / 256, 256, 0, stream>>>(colp, dis, E);
  dis_kernel<<<(n + 255) / 256, 256, 0, stream>>>(dis, n);
  ztx_gemm_kernel<<<(64 * ZSLICES) / 8, 256, 0, stream>>>(YhT, XhT, Zpart, n);
  zreduce_kernel<<<(DIM * DIM) / 256, 256, 0, stream>>>(Zpart, ZhT);
  xw_gemm_kernel<<<tiles / 8, 256, 0, stream>>>(Xh, WhT, xw, tiles);
  scatter_kernel<<<E / 8, 256, 0, stream>>>(rowp, colp, dis, xw, g, E);
  self_bias_kernel<<<(n * DIM) / 256, 256, 0, stream>>>(xw, dis, b, g, n * DIM);
  att_final_kernel<<<tiles / 8, 256, 0, stream>>>(Yh, ZhT, g, out, tiles);
}
